// GCNModel_84404697301756
// MI455X (gfx1250) — compile-verified, matrix operand order checked by reading, not run on hardware
//
#include <hip/hip_runtime.h>
#include <hip/hip_bf16.h>

// ---------------------------------------------------------------------------
// GCN forward on MI455X (gfx1250): bf16 WMMA GEMMs (LDS-staged weights) +
// L2-resident edge scatter.
// ---------------------------------------------------------------------------

typedef __attribute__((ext_vector_type(16))) __bf16 v16bf;
typedef __attribute__((ext_vector_type(8)))  __bf16 v8bf;
typedef __attribute__((ext_vector_type(8)))  float  v8f;

#define NNODES 100000
#define NEDGES 1600000
#define FIN    128
#define H1DIM  128
#define H2DIM  64
#define NCLS   10
#define NGRAPH 64

// ---------------------------------------------------- utility fill / degree
__global__ void fill_f32(float* __restrict__ p, float v, long long n) {
    long long t = (long long)blockIdx.x * blockDim.x + threadIdx.x;
    if (t < n) p[t] = v;
}

__global__ void deg_accum(const int* __restrict__ ei, float* __restrict__ deg, int E) {
    int e = blockIdx.x * blockDim.x + threadIdx.x;
    if (e < E) atomicAdd(&deg[ei[E + e]], 1.0f);   // dst row of edge_index
}

__global__ void rsqrt_ip(float* __restrict__ deg, int n) {
    int t = blockIdx.x * blockDim.x + threadIdx.x;
    if (t < n) deg[t] = rsqrtf(deg[t]);
}

// ---------------------------------------------------- bf16 WMMA GEMM, K=128
// A: [M x 128] f32 row-major, W: [128 x NOUT] f32 row-major, Out: [M x NOUT].
// 8 waves/block (128-row tile). W is staged once per block into LDS as bf16,
// TRANSPOSED ([col][k]) so each lane's B fragment is two contiguous 16-byte
// LDS reads (ds_load_b128). A is read with float4 (global_load_b128) and
// packed to bf16 in-register (v_cvt_pk_bf16_f32).
template <int NOUT>
__global__ void gemm_bf16_wmma(const float* __restrict__ A,
                               const float* __restrict__ W,
                               float* __restrict__ Out, int M) {
    constexpr int K  = 128;
    constexpr int NT = NOUT / 16;

    __shared__ alignas(16) __bf16 wlds[NOUT * K];   // W^T, bf16: 32KB (NOUT=128)

    // ---- cooperative stage + transpose + convert
    for (int i = threadIdx.x; i < NOUT * K; i += blockDim.x) {
        const int col = i / K;
        const int k   = i - col * K;
        wlds[i] = (__bf16)W[(size_t)k * NOUT + col];
    }
    __syncthreads();

    const int wave    = threadIdx.x >> 5;
    const int lane    = threadIdx.x & 31;
    const int lrow    = lane & 15;       // row (A) / col (B) within 16-tile
    const int hi      = lane >> 4;       // half-wave selects K sub-range
    const int rowBase = (blockIdx.x * (blockDim.x >> 5) + wave) * 16;
    const int row     = rowBase + lrow;
    const int arow    = row < M ? row : M - 1;   // clamp; tail rows not stored

    v8f acc[NT] = {};

    for (int k0 = 0; k0 < K; k0 += 32) {
        // ---- A fragment: lane holds row `arow`, K = k0+hi*8+{0..7, 16..23}
        const float4* ap = (const float4*)(A + (size_t)arow * K + k0 + hi * 8);
        const float4 a0 = ap[0];   // K+0..3
        const float4 a1 = ap[1];   // K+4..7
        const float4 a2 = ap[4];   // K+16..19
        const float4 a3 = ap[5];   // K+20..23
        v16bf afrag;
        afrag[0]  = (__bf16)a0.x; afrag[1]  = (__bf16)a0.y;
        afrag[2]  = (__bf16)a0.z; afrag[3]  = (__bf16)a0.w;
        afrag[4]  = (__bf16)a1.x; afrag[5]  = (__bf16)a1.y;
        afrag[6]  = (__bf16)a1.z; afrag[7]  = (__bf16)a1.w;
        afrag[8]  = (__bf16)a2.x; afrag[9]  = (__bf16)a2.y;
        afrag[10] = (__bf16)a2.z; afrag[11] = (__bf16)a2.w;
        afrag[12] = (__bf16)a3.x; afrag[13] = (__bf16)a3.y;
        afrag[14] = (__bf16)a3.z; afrag[15] = (__bf16)a3.w;

        // ---- per 16-column tile: B fragment from LDS + WMMA
#pragma unroll
        for (int nt = 0; nt < NT; ++nt) {
            const int col = nt * 16 + lrow;
            const __bf16* wp = &wlds[col * K + k0 + hi * 8];
            const v8bf blo = *(const v8bf*)(wp);        // K+0..7   (b128)
            const v8bf bhi = *(const v8bf*)(wp + 16);   // K+16..23 (b128)
            const v16bf bfrag = __builtin_shufflevector(
                blo, bhi, 0, 1, 2, 3, 4, 5, 6, 7, 8, 9, 10, 11, 12, 13, 14, 15);

            acc[nt] = __builtin_amdgcn_wmma_f32_16x16x32_bf16(
                false, afrag, false, bfrag, (short)0, acc[nt], false, false);
        }
    }

    // ---- store: C/D layout — VGPR v holds M = v + hi*8, N = lrow
    if (rowBase < M) {
#pragma unroll
        for (int nt = 0; nt < NT; ++nt) {
#pragma unroll
            for (int v = 0; v < 8; ++v) {
                const int r = rowBase + hi * 8 + v;
                if (r < M) Out[(size_t)r * NOUT + nt * 16 + lrow] = acc[nt][v];
            }
        }
    }
}

// ---------------------------------------------------- edge scatter (gather+atomic)
// One thread = one edge x 4 features. h/agg stay resident in the 192MB L2.
template <int H>
__global__ void msg_pass(const int* __restrict__ ei, const float* __restrict__ h,
                         const float* __restrict__ dinv, float* __restrict__ agg, int E) {
    constexpr int NV = H / 4;
    const long long tid   = (long long)blockIdx.x * blockDim.x + threadIdx.x;
    const long long total = (long long)E * NV;
    if (tid >= total) return;
    const int e = (int)(tid / NV);
    const int f = (int)(tid % NV) * 4;

    __builtin_prefetch(&ei[e + 2048], 0, 0);          // global_prefetch_b8

    const int s = ei[e];
    const int d = ei[E + e];
    const float norm = dinv[s] * dinv[d];

    const float4 v = *(const float4*)(h + (size_t)s * H + f);
    float* o = agg + (size_t)d * H + f;
    atomicAdd(o + 0, v.x * norm);
    atomicAdd(o + 1, v.y * norm);
    atomicAdd(o + 2, v.z * norm);
    atomicAdd(o + 3, v.w * norm);
}

// ---------------------------------------------------- self-loop + bias (+ReLU)
template <int H, bool RELU>
__global__ void selfloop_bias(const float* __restrict__ h, const float* __restrict__ dinv,
                              const float* __restrict__ bias, float* __restrict__ agg, int Nn) {
    const long long tid = (long long)blockIdx.x * blockDim.x + threadIdx.x;
    if (tid >= (long long)Nn * H) return;
    const int n = (int)(tid / H);
    const int f = (int)(tid % H);
    const float di = dinv[n];
    float v = agg[tid] + h[tid] * (di * di) + bias[f];
    if (RELU) v = fmaxf(v, 0.0f);
    agg[tid] = v;
}

// ---------------------------------------------------- mean pool accumulation
__global__ void pool_accum(const float* __restrict__ h, const int* __restrict__ batch,
                           float* __restrict__ sums, float* __restrict__ cnt, int Nn) {
    const long long tid = (long long)blockIdx.x * blockDim.x + threadIdx.x;
    if (tid >= (long long)Nn * H2DIM) return;
    const int n = (int)(tid >> 6);
    const int f = (int)(tid & 63);
    const int g = batch[n];
    atomicAdd(&sums[g * H2DIM + f], h[tid]);
    if (f == 0) atomicAdd(&cnt[g], 1.0f);
}

// ---------------------------------------------------- final FC (64x64 @ 64x10)
__global__ void final_fc(const float* __restrict__ sums, const float* __restrict__ cnt,
                         const float* __restrict__ Wfc, const float* __restrict__ bfc,
                         float* __restrict__ out) {
    const int tid = threadIdx.x;
    if (tid >= NGRAPH * NCLS) return;
    const int g = tid / NCLS;
    const int c = tid % NCLS;
    const float inv = 1.0f / fmaxf(cnt[g], 1.0f);
    float acc = bfc[c];
#pragma unroll 8
    for (int f = 0; f < H2DIM; ++f)
        acc += (sums[g * H2DIM + f] * inv) * Wfc[f * NCLS + c];
    out[g * NCLS + c] = acc;
}

// ---------------------------------------------------------------------------
extern "C" void kernel_launch(void* const* d_in, const int* in_sizes, int n_in,
                              void* d_out, int out_size, void* d_ws, size_t ws_size,
                              hipStream_t stream) {
    const float* x     = (const float*)d_in[0];
    const int*   ei    = (const int*)  d_in[1];
    const int*   batch = (const int*)  d_in[2];
    const float* W1    = (const float*)d_in[3];
    const float* b1    = (const float*)d_in[4];
    const float* W2    = (const float*)d_in[5];
    const float* b2    = (const float*)d_in[6];
    const float* Wfc   = (const float*)d_in[7];
    const float* bfc   = (const float*)d_in[8];
    float* out = (float*)d_out;

    // ---- workspace layout: deg/dinv | bufA (N*128) | bufB (N*128) | pool
    char* ws = (char*)d_ws;
    float* deg = (float*)ws;
    const size_t offA = ((size_t)NNODES * 4 + 255) & ~(size_t)255;
    float* bufA = (float*)(ws + offA);                                   // h
    float* bufB = (float*)(ws + offA + (size_t)NNODES * H1DIM * 4);      // agg
    float* sums = (float*)(ws + offA + 2 * (size_t)NNODES * H1DIM * 4);
    float* cnt  = sums + NGRAPH * H2DIM;

    const int B = 256;
    auto blocks = [](long long n, int b) { return (int)((n + b - 1) / b); };

    // 1) degree with self-loop, then rsqrt in place -> dinv
    fill_f32<<<blocks(NNODES, B), B, 0, stream>>>(deg, 1.0f, NNODES);
    deg_accum<<<blocks(NEDGES, B), B, 0, stream>>>(ei, deg, NEDGES);
    rsqrt_ip<<<blocks(NNODES, B), B, 0, stream>>>(deg, NNODES);

    // 2) h1 = x @ W1   (bf16 WMMA, f32 accum)  -> bufA
    gemm_bf16_wmma<H1DIM><<<blocks(NNODES, 128), 256, 0, stream>>>(x, W1, bufA, NNODES);

    // 3) agg1 = scatter(norm * h1[src]) -> bufB
    fill_f32<<<blocks((long long)NNODES * H1DIM, B), B, 0, stream>>>(bufB, 0.0f, (long long)NNODES * H1DIM);
    msg_pass<H1DIM><<<blocks((long long)NEDGES * (H1DIM / 4), B), B, 0, stream>>>(ei, bufA, deg, bufB, NEDGES);

    // 4) bufB = relu(agg1 + h1*dinv^2 + b1)
    selfloop_bias<H1DIM, true><<<blocks((long long)NNODES * H1DIM, B), B, 0, stream>>>(bufA, deg, b1, bufB, NNODES);

    // 5) h2 = bufB @ W2  (K=128 again) -> bufA
    gemm_bf16_wmma<H2DIM><<<blocks(NNODES, 128), 256, 0, stream>>>(bufB, W2, bufA, NNODES);

    // 6) agg2 -> bufB
    fill_f32<<<blocks((long long)NNODES * H2DIM, B), B, 0, stream>>>(bufB, 0.0f, (long long)NNODES * H2DIM);
    msg_pass<H2DIM><<<blocks((long long)NEDGES * (H2DIM / 4), B), B, 0, stream>>>(ei, bufA, deg, bufB, NEDGES);

    // 7) bufB = agg2 + h2*dinv^2 + b2   (no relu)
    selfloop_bias<H2DIM, false><<<blocks((long long)NNODES * H2DIM, B), B, 0, stream>>>(bufA, deg, b2, bufB, NNODES);

    // 8) global mean pool + final FC
    fill_f32<<<blocks(NGRAPH * H2DIM + NGRAPH, B), B, 0, stream>>>(sums, 0.0f, NGRAPH * H2DIM + NGRAPH);
    pool_accum<<<blocks((long long)NNODES * H2DIM, B), B, 0, stream>>>(bufB, batch, sums, cnt, NNODES);
    final_fc<<<1, NGRAPH * NCLS, 0, stream>>>(sums, cnt, Wfc, bfc, out);
}
